// build_model_81355270521251
// MI455X (gfx1250) — compile-verified, matrix operand order checked by reading, not run on hardware
//
#include <hip/hip_runtime.h>
#include <math.h>

// ---- problem dims (fixed by reference) ----
#define Bn   32
#define Hn   48
#define Tn   24
#define Sn   256
#define An   64
#define Rn   8
#define Fn   1024
#define Ln   2
#define SEQn 49
#define EPSn 1e-5f

typedef __attribute__((ext_vector_type(16))) __bf16 v16bf;
typedef __attribute__((ext_vector_type(8)))  __bf16 v8bf;
typedef __attribute__((ext_vector_type(8)))  float  v8f;

// ---------------- device helpers ----------------

// Pack 16 bf16 values per the CDNA5 16-bit A/B lane layout:
// lane-half h holds K = h*8 .. h*8+7 and K = 16+h*8 .. 16+h*8+7.
// Two 16B loads -> global_load_b128 each.
__device__ __forceinline__ v16bf pack16_bf16(const unsigned short* __restrict__ p, int h) {
  const v8bf* q0 = (const v8bf*)(const void*)(p + h * 8);
  const v8bf* q1 = (const v8bf*)(const void*)(p + 16 + h * 8);
  v8bf lo = *q0, hi = *q1;
  return __builtin_shufflevector(lo, hi, 0, 1, 2, 3, 4, 5, 6, 7,
                                 8, 9, 10, 11, 12, 13, 14, 15);
}

__device__ __forceinline__ unsigned short f2bf(float f) {
  unsigned int u = __float_as_uint(f);
  unsigned int r = (u + 0x7fffu + ((u >> 16) & 1u)) >> 16;  // RNE to bf16
  return (unsigned short)r;
}

__device__ __forceinline__ float gelu_exact(float x) {
  return 0.5f * x * (1.0f + erff(x * 0.70710678118654752440f));
}
__device__ __forceinline__ float sigm(float x) { return 1.0f / (1.0f + expf(-x)); }

// ---------------- small utility kernels ----------------

__global__ void cvt_bf16_kernel(const float* __restrict__ src,
                                unsigned short* __restrict__ dst, int n) {
  int i = blockIdx.x * blockDim.x + threadIdx.x;
  if (i < n) dst[i] = f2bf(src[i]);
}

__global__ void zero_f32_kernel(float* __restrict__ p, int n) {
  int i = blockIdx.x * blockDim.x + threadIdx.x;
  if (i < n) p[i] = 0.0f;
}

__global__ void zero_u16_kernel(unsigned short* __restrict__ p, int n) {
  int i = blockIdx.x * blockDim.x + threadIdx.x;
  if (i < n) p[i] = 0;
}

// dst_bf16 = bf16(gelu(src_f32))
__global__ void gelu_cvt_kernel(const float* __restrict__ src,
                                unsigned short* __restrict__ dst, int n) {
  int i = blockIdx.x * blockDim.x + threadIdx.x;
  if (i < n) dst[i] = f2bf(gelu_exact(src[i]));
}

// x_fut(bf16) = bf16(gelu(pres_ln + fa_ln[:, t, :]))
__global__ void combine_fa_kernel(const float* __restrict__ pres,
                                  const float* __restrict__ fa, int t,
                                  unsigned short* __restrict__ xo) {
  int i = blockIdx.x * blockDim.x + threadIdx.x;
  if (i < Bn * Fn) {
    int b = i >> 10, f = i & 1023;
    xo[i] = f2bf(gelu_exact(pres[i] + fa[((size_t)(b * Tn + t)) * Fn + f]));
  }
}

// ---------------- GEMV(+bias)+LayerNorm (one-time / tiny-per-step ops) ----------------
// out row index = (row/per)*strideA + (row%per)*strideB + base  (in floats).
__global__ __launch_bounds__(256)
void gemv_ln_kernel(const float* __restrict__ in, int inStride, int K,
                    const float* __restrict__ W, const float* __restrict__ bias,
                    const float* __restrict__ g, const float* __restrict__ bn,
                    float* __restrict__ out, int per, long strideA, long strideB,
                    long base, int accumulate) {
  __shared__ float xin[256];
  __shared__ float red[256];
  const int row = blockIdx.x, tid = threadIdx.x;
  if (tid < K) xin[tid] = in[(size_t)row * inStride + tid];
  __syncthreads();
  float v[4];
  float partial = 0.0f;
#pragma unroll
  for (int j = 0; j < 4; ++j) {
    const int n = tid + j * 256;
    const float* wr = W + (size_t)n * K;
    float acc = bias[n];
    for (int k = 0; k < K; ++k) acc += xin[k] * wr[k];
    v[j] = acc;
    partial += acc;
  }
  red[tid] = partial;
  __syncthreads();
  for (int s = 128; s > 0; s >>= 1) {
    if (tid < s) red[tid] += red[tid + s];
    __syncthreads();
  }
  const float mean = red[0] * (1.0f / Fn);
  __syncthreads();
  float sq = 0.0f;
#pragma unroll
  for (int j = 0; j < 4; ++j) {
    float d = v[j] - mean;
    sq += d * d;
  }
  red[tid] = sq;
  __syncthreads();
  for (int s = 128; s > 0; s >>= 1) {
    if (tid < s) red[tid] += red[tid + s];
    __syncthreads();
  }
  const float rstd = rsqrtf(red[0] * (1.0f / Fn) + EPSn);
  const long obase = (long)(row / per) * strideA + (long)(row % per) * strideB + base;
#pragma unroll
  for (int j = 0; j < 4; ++j) {
    const int n = tid + j * 256;
    float val = (v[j] - mean) * rstd * g[n] + bn[n];
    if (accumulate) out[obase + n] += val;
    else out[obase + n] = val;
  }
}

// ---------------- plain GEMV + bias (+optional tanh) for heads ----------------
__global__ __launch_bounds__(128)
void gemv_bias_kernel(const float* __restrict__ X, int xld, int K,
                      const float* __restrict__ W, const float* __restrict__ bv,
                      int N, int act, float* __restrict__ out, int outStride,
                      int outBase, float* __restrict__ out2, int nb) {
  __shared__ float xs[1024];
  const int row = blockIdx.x / nb;
  const int n = (blockIdx.x % nb) * 128 + threadIdx.x;
  for (int i = threadIdx.x; i < K; i += 128) xs[i] = X[(size_t)row * xld + i];
  __syncthreads();
  if (n < N) {
    const float* wr = W + (size_t)n * K;
    float acc = bv[n];
    for (int k = 0; k < K; k += 4)
      acc += xs[k] * wr[k] + xs[k + 1] * wr[k + 1] + xs[k + 2] * wr[k + 2] +
             xs[k + 3] * wr[k + 3];
    if (act) acc = tanhf(acc);
    out[(size_t)row * outStride + outBase + n] = acc;
    if (out2) out2[(size_t)row * N + n] = acc;
  }
}

// ---------------- core: one GRU layer step, fully fused (WMMA bf16) ----------------
// grid.x = 128: blockIdx>>1 = f-tile (16 cols of F), blockIdx&1 = m-tile (16 rows of B).
// 24 waves: q in 0..5 selects {gi_r, gi_z, gi_n, gh_r, gh_z, gh_n}; ks in 0..3 splits K=1024.
// Activations arrive pre-converted to bf16, so the WMMA input chain is pure b128 loads.
__global__ __launch_bounds__(768)
void gru_layer_kernel(const unsigned short* __restrict__ x_bf, int xld,
                      const float* __restrict__ h_in_f32,
                      const unsigned short* __restrict__ h_in_bf,
                      float* __restrict__ h_out_f32,
                      unsigned short* __restrict__ h_out_bf,
                      const unsigned short* __restrict__ Wi,
                      const unsigned short* __restrict__ Wh,
                      const float* __restrict__ bi, const float* __restrict__ bh) {
  __shared__ float tiles[6][4][16][16];
  const int tid = threadIdx.x;
  const int f0 = (blockIdx.x >> 1) * 16;
  const int m = blockIdx.x & 1;
  const int w = tid >> 5, lane = tid & 31;
  const int q = w % 6, ks = w / 6;
  {
    const bool useH = (q >= 3);
    const int gate = useH ? (q - 3) : q;
    const unsigned short* Ab = useH ? h_in_bf : x_bf;
    const int lda = useH ? Fn : xld;
    const unsigned short* Wb = useH ? Wh : Wi;
    const int half = lane >> 4;
    const unsigned short* arow = Ab + (size_t)(m * 16 + (lane & 15)) * lda + ks * 256;
    const unsigned short* brow =
        Wb + (size_t)(gate * Fn + f0 + (lane & 15)) * Fn + ks * 256;
    v8f acc = {0.f, 0.f, 0.f, 0.f, 0.f, 0.f, 0.f, 0.f};
#pragma unroll
    for (int k0 = 0; k0 < 256; k0 += 32) {
      v16bf a = pack16_bf16(arow + k0, half);
      v16bf b = pack16_bf16(brow + k0, half);
      // D(f32) = A(bf16) x B(bf16) + C(f32): v_wmma_f32_16x16x32_bf16
      acc = __builtin_amdgcn_wmma_f32_16x16x32_bf16(false, a, false, b, (short)0,
                                                    acc, false, false);
    }
    const int n = lane & 15;
#pragma unroll
    for (int r = 0; r < 8; ++r) tiles[q][ks][r + 8 * half][n] = acc[r];
  }
  __syncthreads();
  if (tid < 256) {
    const int lr = tid >> 4, col = tid & 15;
    float s0 = 0, s1 = 0, s2 = 0, s3 = 0, s4 = 0, s5 = 0;
#pragma unroll
    for (int k = 0; k < 4; ++k) {
      s0 += tiles[0][k][lr][col];
      s1 += tiles[1][k][lr][col];
      s2 += tiles[2][k][lr][col];
      s3 += tiles[3][k][lr][col];
      s4 += tiles[4][k][lr][col];
      s5 += tiles[5][k][lr][col];
    }
    const int fc = f0 + col;
    const float ir = s0 + bi[fc], iz = s1 + bi[Fn + fc], inn = s2 + bi[2 * Fn + fc];
    const float hr = s3 + bh[fc], hz = s4 + bh[Fn + fc], hn = s5 + bh[2 * Fn + fc];
    const float rg = sigm(ir + hr);
    const float zg = sigm(iz + hz);
    const float ng = tanhf(inn + rg * hn);
    const int row = m * 16 + lr;
    const size_t idx = (size_t)row * Fn + fc;
    const float hold = h_in_f32[idx];
    const float hnew = (1.0f - zg) * ng + zg * hold;
    h_out_f32[idx] = hnew;
    h_out_bf[idx] = f2bf(hnew);
  }
}

// ---------------- host orchestration ----------------

extern "C" void kernel_launch(void* const* d_in, const int* in_sizes, int n_in,
                              void* d_out, int out_size, void* d_ws, size_t ws_size,
                              hipStream_t stream) {
  (void)in_sizes; (void)n_in; (void)out_size; (void)ws_size;
  const float* history_s = (const float*)d_in[0];
  const float* history_a = (const float*)d_in[1];
  const float* present_s = (const float*)d_in[2];
  /* d_in[3] future_s: unused by the reference computation */
  const float* future_a  = (const float*)d_in[4];
  const float* W_state   = (const float*)d_in[5];
  const float* b_state   = (const float*)d_in[6];
  const float* g_sn      = (const float*)d_in[7];
  const float* b_sn      = (const float*)d_in[8];
  const float* W_action  = (const float*)d_in[9];
  const float* b_action  = (const float*)d_in[10];
  const float* g_an      = (const float*)d_in[11];
  const float* b_an      = (const float*)d_in[12];
  const float* gru_Wi    = (const float*)d_in[13];
  const float* gru_Wh    = (const float*)d_in[14];
  const float* gru_bi    = (const float*)d_in[15];
  const float* gru_bh    = (const float*)d_in[16];
  const float* W_reward  = (const float*)d_in[17];
  const float* b_reward  = (const float*)d_in[18];
  const float* W_sout    = (const float*)d_in[19];
  const float* b_sout    = (const float*)d_in[20];

  float* out_r = (float*)d_out;                 // [B,T,R]
  float* out_s = (float*)d_out + Bn * Tn * Rn;  // [B,T,S]

  // workspace carve-up (~40 MB): bf16 GRU weights stay L2-resident across all 72 steps
  char* ws = (char*)d_ws;
  size_t off = 0;
  auto take = [&](size_t bytes) {
    char* p = ws + off;
    off += (bytes + 255) & ~(size_t)255;
    return p;
  };
  unsigned short* wsWi   = (unsigned short*)take((size_t)Ln * 3 * Fn * Fn * 2);
  unsigned short* wsWh   = (unsigned short*)take((size_t)Ln * 3 * Fn * Fn * 2);
  float*          seq0f  = (float*)take((size_t)Bn * SEQn * Fn * 4);   // pre-gelu f32
  unsigned short* seq0b  = (unsigned short*)take((size_t)Bn * SEQn * Fn * 2);  // gelu'd bf16
  float*          fa_ln  = (float*)take((size_t)Bn * Tn * Fn * 4);
  float*          hbf32  = (float*)take((size_t)Ln * 2 * Bn * Fn * 4);  // h double-buffers f32
  unsigned short* hbbf   = (unsigned short*)take((size_t)Ln * 2 * Bn * Fn * 2);  // h bf16 mirror
  unsigned short* x_fut  = (unsigned short*)take((size_t)Bn * Fn * 2);
  float*          s_buf  = (float*)take((size_t)Bn * Sn * 4);
  float*          pres_ln= (float*)take((size_t)Bn * Fn * 4);

  const int nW = Ln * 3 * Fn * Fn;
  cvt_bf16_kernel<<<(nW + 255) / 256, 256, 0, stream>>>(gru_Wi, wsWi, nW);
  cvt_bf16_kernel<<<(nW + 255) / 256, 256, 0, stream>>>(gru_Wh, wsWh, nW);
  const int nH = Ln * 2 * Bn * Fn;
  zero_f32_kernel<<<(nH + 255) / 256, 256, 0, stream>>>(hbf32, nH);
  zero_u16_kernel<<<(nH + 255) / 256, 256, 0, stream>>>(hbbf, nH);

  // embeddings + LN; seq0[b,t<48] = LN(state)+LN(action); seq0[b,48] = LN(present)+LN(fa[0])
  gemv_ln_kernel<<<Bn * Hn, 256, 0, stream>>>(history_s, Sn, Sn, W_state, b_state,
                                              g_sn, b_sn, seq0f, Hn, (long)SEQn * Fn,
                                              (long)Fn, 0, 0);
  gemv_ln_kernel<<<Bn * Hn, 256, 0, stream>>>(history_a, An, An, W_action, b_action,
                                              g_an, b_an, seq0f, Hn, (long)SEQn * Fn,
                                              (long)Fn, 0, 1);
  gemv_ln_kernel<<<Bn, 256, 0, stream>>>(present_s, Sn, Sn, W_state, b_state, g_sn,
                                         b_sn, seq0f, 1, (long)SEQn * Fn, (long)Fn,
                                         (long)Hn * Fn, 0);
  gemv_ln_kernel<<<Bn, 256, 0, stream>>>(future_a, Tn * An, An, W_action, b_action,
                                         g_an, b_an, seq0f, 1, (long)SEQn * Fn,
                                         (long)Fn, (long)Hn * Fn, 1);
  gemv_ln_kernel<<<Bn * Tn, 256, 0, stream>>>(future_a, An, An, W_action, b_action,
                                              g_an, b_an, fa_ln, Bn * Tn, 0, (long)Fn,
                                              0, 0);
  const int nSeq = Bn * SEQn * Fn;
  gelu_cvt_kernel<<<(nSeq + 255) / 256, 256, 0, stream>>>(seq0f, seq0b, nSeq);

  auto hf = [&](int l, int p) { return hbf32 + (size_t)(l * 2 + p) * Bn * Fn; };
  auto hb = [&](int l, int p) { return hbbf + (size_t)(l * 2 + p) * Bn * Fn; };
  int par = 0;

  // 49 sequential steps over history+present window
  for (int k = 0; k < SEQn; ++k) {
    gru_layer_kernel<<<128, 768, 0, stream>>>(
        seq0b + (size_t)k * Fn, SEQn * Fn, hf(0, par), hb(0, par), hf(0, 1 - par),
        hb(0, 1 - par), wsWi, wsWh, gru_bi, gru_bh);
    gru_layer_kernel<<<128, 768, 0, stream>>>(
        hb(0, 1 - par), Fn, hf(1, par), hb(1, par), hf(1, 1 - par), hb(1, 1 - par),
        wsWi + (size_t)3 * Fn * Fn, wsWh + (size_t)3 * Fn * Fn, gru_bi + 3 * Fn,
        gru_bh + 3 * Fn);
    par ^= 1;
  }

  // future rollout: head(t) -> pres LN -> gelu(+fa) -> GRU step -> head(t+1) ...
  for (int t = 0; t < Tn; ++t) {
    if (t > 0) {
      gemv_ln_kernel<<<Bn, 256, 0, stream>>>(s_buf, Sn, Sn, W_state, b_state, g_sn,
                                             b_sn, pres_ln, Bn, 0, (long)Fn, 0, 0);
      combine_fa_kernel<<<(Bn * Fn + 255) / 256, 256, 0, stream>>>(pres_ln, fa_ln, t,
                                                                   x_fut);
      gru_layer_kernel<<<128, 768, 0, stream>>>(
          x_fut, Fn, hf(0, par), hb(0, par), hf(0, 1 - par), hb(0, 1 - par), wsWi,
          wsWh, gru_bi, gru_bh);
      gru_layer_kernel<<<128, 768, 0, stream>>>(
          hb(0, 1 - par), Fn, hf(1, par), hb(1, par), hf(1, 1 - par), hb(1, 1 - par),
          wsWi + (size_t)3 * Fn * Fn, wsWh + (size_t)3 * Fn * Fn, gru_bi + 3 * Fn,
          gru_bh + 3 * Fn);
      par ^= 1;
    }
    const float* y = hf(1, par);  // layer-1 output of latest step (f32)
    gemv_bias_kernel<<<Bn * 2, 128, 0, stream>>>(y, Fn, Fn, W_sout, b_sout, Sn, 0,
                                                 out_s, Tn * Sn, t * Sn, s_buf, 2);
    gemv_bias_kernel<<<Bn, 128, 0, stream>>>(y, Fn, Fn, W_reward, b_reward, Rn, 1,
                                             out_r, Tn * Rn, t * Rn, (float*)0, 1);
  }
}
